// NodeClassifier_42073499631700
// MI455X (gfx1250) — matrix operand order repaired and checked by reading, not compiled, not run on hardware
//
#include <hip/hip_runtime.h>
#include <hip/hip_bf16.h>

typedef __attribute__((ext_vector_type(16))) _Float16 v16h;
typedef __attribute__((ext_vector_type(8)))  _Float16 v8h;
typedef __attribute__((ext_vector_type(8)))  float    v8f;

// ---------------- small utility kernels ----------------

__global__ void k_fill_f32(float* __restrict__ p, float v, int n) {
  int i = blockIdx.x * blockDim.x + threadIdx.x;
  if (i < n) p[i] = v;
}

__global__ void k_fill_from(float* __restrict__ p, const float* __restrict__ v, int n) {
  int i = blockIdx.x * blockDim.x + threadIdx.x;
  if (i < n) p[i] = v[0];
}

__global__ void k_deg_accum(const int* __restrict__ dst, float* __restrict__ deg, int e) {
  int i = blockIdx.x * blockDim.x + threadIdx.x;
  if (i < e) atomicAdd(&deg[dst[i]], 1.0f);
}

__global__ void k_rsqrt(float* __restrict__ d, int n) {
  int i = blockIdx.x * blockDim.x + threadIdx.x;
  if (i < n) d[i] = rsqrtf(d[i]);   // deg >= 1 always (self-loops), no zero guard needed
}

__global__ void k_cvt_f16(const float* __restrict__ x, _Float16* __restrict__ xh, int n) {
  int i = blockIdx.x * blockDim.x + threadIdx.x;
  if (i < n) xh[i] = (_Float16)x[i];
}

// W1 [K=in_ch][N=hid] row-major f32  ->  W1T [N][K] f16 (contiguous K per column)
__global__ void k_w1t(const float* __restrict__ W1, _Float16* __restrict__ w1t, int K, int N) {
  int t = blockIdx.x * blockDim.x + threadIdx.x;
  if (t < K * N) {
    int nn = t / K, kk = t % K;
    w1t[t] = (_Float16)W1[kk * N + nn];
  }
}

// ---------------- GEMM1: h[100000x32] = x[100000x128] @ W1 via WMMA ----------------
// One wave per 16-row M tile; two 16-col N tiles; K unrolled 128 = 4 x 32.
// Fragment packing used here (wave32, same contiguous K order for A and B):
//   A 16x32 f16 : lane L holds row m=L&15; element j -> K = (L>>4)*16 + j
//   B 32x16 f16 : lane L holds col n=L&15; element j -> K = (L>>4)*16 + j
//   C/D 16x16 f32: element r -> row m = r + (L>>4)*8, col n = L&15
__global__ void k_gemm1_wmma(const _Float16* __restrict__ xh,
                             const _Float16* __restrict__ w1t,
                             float* __restrict__ h, int n_nodes) {
  const int lane = threadIdx.x & 31;
  const int wave = (blockIdx.x * blockDim.x + threadIdx.x) >> 5;
  const int m0 = wave * 16;
  if (m0 >= n_nodes) return;              // wave-uniform exit; WMMA never reached with EXEC==0
  const int mlo = lane & 15;
  const int hi  = lane >> 4;
  const size_t aRow = (size_t)(m0 + mlo) * 128;

  v8f c0 = {}; v8f c1 = {};
#pragma unroll
  for (int kb = 0; kb < 128; kb += 32) {
    // A fragment: two contiguous 16B loads per lane
    const _Float16* ap = xh + aRow + kb + hi * 16;
    v8h alo = *(const v8h*)(ap);          // elements 0..7  : K = kb + hi*16 + 0..7
    v8h ahi = *(const v8h*)(ap + 8);      // elements 8..15 : K = kb + hi*16 + 8..15
    v16h a;
#pragma unroll
    for (int j = 0; j < 8; ++j) { a[j] = alo[j]; a[j + 8] = ahi[j]; }

    // B fragments from transposed W1: 16 contiguous halves per lane
    const _Float16* bp0 = w1t + (size_t)mlo * 128 + kb + hi * 16;          // N-tile 0
    const _Float16* bp1 = bp0 + (size_t)16 * 128;                           // N-tile 1
    v16h b0 = *(const v16h*)bp0;
    v16h b1 = *(const v16h*)bp1;

    c0 = __builtin_amdgcn_wmma_f32_16x16x32_f16(false, a, false, b0, (short)0, c0, false, false);
    c1 = __builtin_amdgcn_wmma_f32_16x16x32_f16(false, a, false, b1, (short)0, c1, false, false);
  }

  float* hrow = h + (size_t)(m0 + hi * 8) * 32 + mlo;
#pragma unroll
  for (int r = 0; r < 8; ++r) {
    hrow[(size_t)r * 32]      = c0[r];
    hrow[(size_t)r * 32 + 16] = c1[r];
  }
}

// ---------------- layer-1 aggregation: one wave per edge, lane = channel ----------------
__global__ void k_agg1(const float* __restrict__ h,
                       const int* __restrict__ src, const int* __restrict__ dst,
                       const float* __restrict__ dis,
                       float* __restrict__ out1, int n_edges, int n_nodes) {
  int t = blockIdx.x * blockDim.x + threadIdx.x;
  int e = t >> 5;
  int c = t & 31;
  int etot = n_edges + n_nodes;
  if (e >= etot) return;
  int s, d;
  if (e < n_edges) { s = src[e]; d = dst[e]; } else { s = e - n_edges; d = s; }
  float w = dis[s] * dis[d];
  atomicAdd(&out1[(size_t)d * 32 + c], h[(size_t)s * 32 + c] * w);
}

// ---------------- bias + relu + (32 -> 1) matvec, one wave per node ----------------
__global__ void k_relu_dot(const float* __restrict__ out1,
                           const float* __restrict__ b1,
                           const float* __restrict__ W2,
                           float* __restrict__ h2, int n_nodes) {
  int t = blockIdx.x * blockDim.x + threadIdx.x;
  int i = t >> 5;
  int c = t & 31;
  if (i >= n_nodes) return;
  float v = fmaxf(out1[(size_t)i * 32 + c] + b1[c], 0.0f) * W2[c];
#pragma unroll
  for (int off = 16; off > 0; off >>= 1) v += __shfl_xor(v, off, 32);
  if (c == 0) h2[i] = v;
}

// ---------------- layer-2 aggregation: one thread per edge ----------------
__global__ void k_agg2(const float* __restrict__ h2,
                       const int* __restrict__ src, const int* __restrict__ dst,
                       const float* __restrict__ dis,
                       float* __restrict__ out, int n_edges, int n_nodes) {
  int e = blockIdx.x * blockDim.x + threadIdx.x;
  int etot = n_edges + n_nodes;
  if (e >= etot) return;
  int s, d;
  if (e < n_edges) { s = src[e]; d = dst[e]; } else { s = e - n_edges; d = s; }
  atomicAdd(&out[d], h2[s] * dis[s] * dis[d]);
}

// ---------------- launch ----------------

extern "C" void kernel_launch(void* const* d_in, const int* in_sizes, int n_in,
                              void* d_out, int out_size, void* d_ws, size_t ws_size,
                              hipStream_t stream) {
  const float* x   = (const float*)d_in[0];
  const int*   ei  = (const int*)d_in[1];
  const float* W1  = (const float*)d_in[2];
  const float* b1  = (const float*)d_in[3];
  const float* W2  = (const float*)d_in[4];
  const float* b2p = (const float*)d_in[5];
  float* out = (float*)d_out;

  const int in_ch = 128, hid = 32;
  const int n = in_sizes[0] / in_ch;   // 100000
  const int E = in_sizes[1] / 2;       // 3200000
  const int* src = ei;
  const int* dst = ei + E;

  // workspace carve-up, 256B aligned
  char* ws = (char*)d_ws;
  size_t off = 0;
  auto carve = [&](size_t bytes) -> void* {
    void* p = ws + off;
    off = (off + bytes + 255) & ~(size_t)255;
    return p;
  };
  float*    dis = (float*)   carve((size_t)n * 4);
  _Float16* xh  = (_Float16*)carve((size_t)n * in_ch * 2);
  _Float16* w1t = (_Float16*)carve((size_t)in_ch * hid * 2);
  float*    h   = (float*)   carve((size_t)n * hid * 4);
  float*    o1  = (float*)   carve((size_t)n * hid * 4);
  float*    h2  = (float*)   carve((size_t)n * 4);
  (void)ws_size; (void)n_in; (void)out_size;

  const int B = 256;
  auto cdiv = [](long long a, long long b) { return (int)((a + b - 1) / b); };

  // 1) normalization coefficients: deg (init 1.0 for self-loop) -> rsqrt in place
  k_fill_f32 <<<cdiv(n, B), B, 0, stream>>>(dis, 1.0f, n);
  k_deg_accum<<<cdiv(E, B), B, 0, stream>>>(dst, dis, E);
  k_rsqrt    <<<cdiv(n, B), B, 0, stream>>>(dis, n);

  // 2) f16 staging for WMMA
  k_cvt_f16<<<cdiv((long long)n * in_ch, B), B, 0, stream>>>(x, xh, n * in_ch);
  k_w1t    <<<cdiv(in_ch * hid, B), B, 0, stream>>>(W1, w1t, in_ch, hid);

  // 3) h = x @ W1 (WMMA), one wave per 16-row tile
  long long tiles = (n + 15) / 16;
  k_gemm1_wmma<<<cdiv(tiles * 32, B), B, 0, stream>>>(xh, w1t, h, n);

  // 4) layer-1 normalized scatter-add
  k_fill_f32<<<cdiv((long long)n * hid, B), B, 0, stream>>>(o1, 0.0f, n * hid);
  k_agg1<<<cdiv((long long)(E + n) * 32, B), B, 0, stream>>>(h, src, dst, dis, o1, E, n);

  // 5) bias + relu + second linear (32 -> 1)
  k_relu_dot<<<cdiv((long long)n * 32, B), B, 0, stream>>>(o1, b1, W2, h2, n);

  // 6) layer-2 scatter-add into output (pre-initialized to b2)
  k_fill_from<<<cdiv(n, B), B, 0, stream>>>(out, b2p, n);
  k_agg2<<<cdiv((long long)(E + n), B), B, 0, stream>>>(h2, src, dst, dis, out, E, n);
}
